// GatedGCNConv_71906342470118
// MI455X (gfx1250) — compile-verified, matrix operand order checked by reading, not run on hardware
//
#include <hip/hip_runtime.h>

#define D 128
#define WPB 8           // wave tiles (16 rows each) per 256-thread block in GEMM
#define EPS_BN   1e-5f
#define EPS_GATE 1e-6f

typedef __attribute__((ext_vector_type(16))) __bf16 v16bf;
typedef __attribute__((ext_vector_type(8)))  float  v8f;

// ---------- helpers ----------
static __device__ __forceinline__ __bf16 f2bf(float x) {
  unsigned u = __builtin_bit_cast(unsigned, x);
  unsigned r = u + 0x7FFFu + ((u >> 16) & 1u);     // round-to-nearest-even
  unsigned short hb = (unsigned short)(r >> 16);
  return __builtin_bit_cast(__bf16, hb);
}
static __device__ __forceinline__ float bf2f(__bf16 b) {
  unsigned short hb = __builtin_bit_cast(unsigned short, b);
  unsigned u = ((unsigned)hb) << 16;
  return __builtin_bit_cast(float, u);
}
static __device__ __forceinline__ void atomAddF(float* p, float v) {
  __hip_atomic_fetch_add(p, v, __ATOMIC_RELAXED, __HIP_MEMORY_SCOPE_AGENT);
}
static __device__ __forceinline__ float4 ld4(const float* p) { return *(const float4*)p; }
static __device__ __forceinline__ void st4(float* p, float4 v) { *(float4*)p = v; }

// =====================================================================
// bf16x3 split-precision GEMM: Y[w] = X @ W[w] + bias[w]   (w < nW <= 3)
// One wave computes a 16 x 128 output stripe. W staged in LDS in
// B-fragment order (hi/lo split). A-fragments loaded once, reused per W.
// ACC is a compile-time flag: 0 -> plain store, 1 -> Y += result.
// =====================================================================
template <int ACC>
__global__ __launch_bounds__(256) void gemm_bf16x3(
    const float* __restrict__ X, int nrows,
    const float* __restrict__ W0, const float* __restrict__ W1, const float* __restrict__ W2,
    const float* __restrict__ b0, const float* __restrict__ b1, const float* __restrict__ b2,
    float* __restrict__ Y0, float* __restrict__ Y1, float* __restrict__ Y2,
    int nW)
{
  __shared__ __align__(32) __bf16 sWhi[D * D];
  __shared__ __align__(32) __bf16 sWlo[D * D];

  const int tid    = threadIdx.x;
  const int lane   = tid & 31;
  const int wave   = tid >> 5;
  const int laneHi = (lane >> 4) & 1;        // lanes 16..31 hold the +8 K slice
  const int mrow   = lane & 15;
  const int rowbase = (blockIdx.x * WPB + wave) * 16;
  const bool fullTile = (rowbase + 16) <= nrows;   // wave-uniform

  // ---- A fragments: 16x32 bf16 layout, hi/lo split of fp32 rows ----
  // Per lane: 4 chunks x two contiguous 8-float runs (global_load_b128 x16).
  v16bf ahi[4], alo[4];
  {
    const int row = rowbase + mrow;
    const bool ok = row < nrows;
    const float* xr = X + (size_t)row * D + (laneHi << 3);
    const float4 z4 = {0.0f, 0.0f, 0.0f, 0.0f};
#pragma unroll
    for (int kt = 0; kt < 4; ++kt) {
      float4 p0 = ok ? ld4(xr + kt * 32 + 0)  : z4;
      float4 p1 = ok ? ld4(xr + kt * 32 + 4)  : z4;
      float4 p2 = ok ? ld4(xr + kt * 32 + 16) : z4;
      float4 p3 = ok ? ld4(xr + kt * 32 + 20) : z4;
      float vals[16] = {p0.x, p0.y, p0.z, p0.w, p1.x, p1.y, p1.z, p1.w,
                        p2.x, p2.y, p2.z, p2.w, p3.x, p3.y, p3.z, p3.w};
#pragma unroll
      for (int h = 0; h < 16; ++h) {
        __bf16 hi = f2bf(vals[h]);
        ahi[kt][h] = hi;
        alo[kt][h] = f2bf(vals[h] - bf2f(hi));
      }
    }
  }

  const float* Ws[3] = {W0, W1, W2};
  const float* Bs[3] = {b0, b1, b2};
  float*       Ysel[3] = {Y0, Y1, Y2};

  for (int widx = 0; widx < nW; ++widx) {
    __syncthreads();
    // ---- stage W[widx] into LDS, permuted into B-fragment lane order ----
    const float* Wm = Ws[widx];
    for (int i = tid; i < D * D; i += 256) {
      int k = i >> 7, n = i & 127;
      float w = Wm[i];
      __bf16 hi = f2bf(w);
      __bf16 lo = f2bf(w - bf2f(hi));
      int kt = k >> 5, kk = k & 31, nt = n >> 4, col = n & 15;
      int ln = col + (((kk >> 3) & 1) << 4);
      int hh = (kk & 7) | ((kk & 16) ? 8 : 0);
      int di = ((((kt << 3) + nt) << 5) | ln) << 4 | hh;
      sWhi[di] = hi;
      sWlo[di] = lo;
    }
    __syncthreads();

    float* Y = Ysel[widx];
    const float* bm = Bs[widx];

    // Bias row for this weight, loaded once (constant-indexed after unroll).
    float bcolArr[8];
    if (bm) {
#pragma unroll
      for (int nt = 0; nt < 8; ++nt) bcolArr[nt] = bm[nt * 16 + mrow];
    } else {
#pragma unroll
      for (int nt = 0; nt < 8; ++nt) bcolArr[nt] = 0.0f;
    }

#pragma unroll
    for (int nt = 0; nt < 8; ++nt) {
      v8f acc;
#pragma unroll
      for (int r = 0; r < 8; ++r) acc[r] = bcolArr[nt];

#pragma unroll
      for (int kt = 0; kt < 4; ++kt) {
        int foff = ((((kt << 3) + nt) << 5) | lane) << 4;
        v16bf bhi = *(const v16bf*)(&sWhi[foff]);
        v16bf blo = *(const v16bf*)(&sWlo[foff]);
        acc = __builtin_amdgcn_wmma_f32_16x16x32_bf16(false, ahi[kt], false, bhi,
                                                      (short)0, acc, false, false);
        acc = __builtin_amdgcn_wmma_f32_16x16x32_bf16(false, ahi[kt], false, blo,
                                                      (short)0, acc, false, false);
        acc = __builtin_amdgcn_wmma_f32_16x16x32_bf16(false, alo[kt], false, bhi,
                                                      (short)0, acc, false, false);
      }

      const int colOut = nt * 16 + mrow;
      float* yb = Y + (size_t)(rowbase + laneHi * 8) * D + colOut;
      if (fullTile) {
        // Fast path: straight-line stores, immediate row offsets.
        if (ACC) {
#pragma unroll
          for (int r = 0; r < 8; ++r) yb[r * D] += acc[r];
        } else {
#pragma unroll
          for (int r = 0; r < 8; ++r) yb[r * D] = acc[r];
        }
      } else {
#pragma unroll
        for (int r = 0; r < 8; ++r) {
          int row = rowbase + r + laneHi * 8;
          if (row < nrows) {
            if (ACC) yb[r * D] += acc[r];
            else     yb[r * D] = acc[r];
          }
        }
      }
    }
  }
}

// =====================================================================
// Edge assembly: epre = Ah[src] + Ah[dst] + Be + Cu[a2g[src]]
// Fused BN column stats (fixed column ownership per thread).
// One warp per edge; lane owns 4 columns.
// =====================================================================
__global__ __launch_bounds__(256) void edge_assemble(
    const float* __restrict__ Ah, const float* __restrict__ Be,
    const float* __restrict__ Cu,
    const int* __restrict__ src, const int* __restrict__ dst,
    const int* __restrict__ a2g,
    float* __restrict__ epre, float* bnsum, float* bnsq, int E)
{
  const int lane = threadIdx.x & 31;
  const int c0 = lane * 4;
  int warp = (blockIdx.x * blockDim.x + threadIdx.x) >> 5;
  int nwarp = (gridDim.x * blockDim.x) >> 5;
  float s0 = 0, s1 = 0, s2 = 0, s3 = 0, q0 = 0, q1 = 0, q2 = 0, q3 = 0;
  for (int j = warp; j < E; j += nwarp) {
    int si = src[j], di = dst[j];
    int g = a2g[si];
    float4 a = ld4(Ah + (size_t)si * D + c0);
    float4 bd = ld4(Ah + (size_t)di * D + c0);
    float4 be = ld4(Be + (size_t)j * D + c0);
    float4 cu = ld4(Cu + (size_t)g * D + c0);
    float4 v;
    v.x = a.x + bd.x + be.x + cu.x;
    v.y = a.y + bd.y + be.y + cu.y;
    v.z = a.z + bd.z + be.z + cu.z;
    v.w = a.w + bd.w + be.w + cu.w;
    st4(epre + (size_t)j * D + c0, v);
    s0 += v.x; s1 += v.y; s2 += v.z; s3 += v.w;
    q0 += v.x * v.x; q1 += v.y * v.y; q2 += v.z * v.z; q3 += v.w * v.w;
  }
  atomAddF(bnsum + c0 + 0, s0); atomAddF(bnsum + c0 + 1, s1);
  atomAddF(bnsum + c0 + 2, s2); atomAddF(bnsum + c0 + 3, s3);
  atomAddF(bnsq + c0 + 0, q0);  atomAddF(bnsq + c0 + 1, q1);
  atomAddF(bnsq + c0 + 2, q2);  atomAddF(bnsq + c0 + 3, q3);
}

// =====================================================================
// BN finalize: scale = gamma*rsqrt(var+eps), shift = beta - mu*scale
// =====================================================================
__global__ void bn_finalize(const float* sum, const float* sq,
                            const float* gamma, const float* beta,
                            float inv_n, float* scale, float* shift)
{
  int c = threadIdx.x;
  float mu = sum[c] * inv_n;
  float var = sq[c] * inv_n - mu * mu;
  float s = gamma[c] * rsqrtf(var + EPS_BN);
  scale[c] = s;
  shift[c] = beta[c] - mu * s;
}

// =====================================================================
// Edge apply: e_new = relu(bn(epre)); sig = sigmoid(e_new);
// scatter num[dst] += sig*Eh[src], den[dst] += sig, S_e[g(dst)] += e_new,
// ecnt[g(dst)] += 1.
// =====================================================================
__global__ __launch_bounds__(256) void edge_apply(
    const float* __restrict__ epre, const float* __restrict__ scale,
    const float* __restrict__ shift, const float* __restrict__ Eh,
    const int* __restrict__ src, const int* __restrict__ dst,
    const int* __restrict__ a2g,
    float* __restrict__ eout, float* num, float* den, float* S_e,
    float* ecnt, int E)
{
  const int lane = threadIdx.x & 31;
  const int c0 = lane * 4;
  int warp = (blockIdx.x * blockDim.x + threadIdx.x) >> 5;
  int nwarp = (gridDim.x * blockDim.x) >> 5;
  float4 sc = ld4(scale + c0);
  float4 sh = ld4(shift + c0);
  for (int j = warp; j < E; j += nwarp) {
    int si = src[j], di = dst[j];
    int g = a2g[di];
    float4 x = ld4(epre + (size_t)j * D + c0);
    float4 y;
    y.x = fmaxf(0.0f, x.x * sc.x + sh.x);
    y.y = fmaxf(0.0f, x.y * sc.y + sh.y);
    y.z = fmaxf(0.0f, x.z * sc.z + sh.z);
    y.w = fmaxf(0.0f, x.w * sc.w + sh.w);
    st4(eout + (size_t)j * D + c0, y);
    float4 sg;
    sg.x = 1.0f / (1.0f + __expf(-y.x));
    sg.y = 1.0f / (1.0f + __expf(-y.y));
    sg.z = 1.0f / (1.0f + __expf(-y.z));
    sg.w = 1.0f / (1.0f + __expf(-y.w));
    float4 eh = ld4(Eh + (size_t)si * D + c0);
    size_t nb = (size_t)di * D + c0;
    atomAddF(num + nb + 0, sg.x * eh.x); atomAddF(num + nb + 1, sg.y * eh.y);
    atomAddF(num + nb + 2, sg.z * eh.z); atomAddF(num + nb + 3, sg.w * eh.w);
    atomAddF(den + nb + 0, sg.x); atomAddF(den + nb + 1, sg.y);
    atomAddF(den + nb + 2, sg.z); atomAddF(den + nb + 3, sg.w);
    size_t gb = (size_t)g * D + c0;
    atomAddF(S_e + gb + 0, y.x); atomAddF(S_e + gb + 1, y.y);
    atomAddF(S_e + gb + 2, y.z); atomAddF(S_e + gb + 3, y.w);
    if (lane == 0) atomAddF(ecnt + g, 1.0f);
  }
}

// =====================================================================
// Atom assembly: hpre = Dh + num/(den+eps) + Fu[a2g]; fused BN stats.
// =====================================================================
__global__ __launch_bounds__(256) void atom_assemble(
    const float* __restrict__ Dh, const float* __restrict__ num,
    const float* __restrict__ den, const float* __restrict__ Fu,
    const int* __restrict__ a2g,
    float* __restrict__ hpre, float* bnsum, float* bnsq, int N)
{
  const int lane = threadIdx.x & 31;
  const int c0 = lane * 4;
  int warp = (blockIdx.x * blockDim.x + threadIdx.x) >> 5;
  int nwarp = (gridDim.x * blockDim.x) >> 5;
  float s0 = 0, s1 = 0, s2 = 0, s3 = 0, q0 = 0, q1 = 0, q2 = 0, q3 = 0;
  for (int i = warp; i < N; i += nwarp) {
    int g = a2g[i];
    size_t ib = (size_t)i * D + c0;
    float4 d = ld4(Dh + ib);
    float4 n4 = ld4(num + ib);
    float4 e4 = ld4(den + ib);
    float4 f = ld4(Fu + (size_t)g * D + c0);
    float4 v;
    v.x = d.x + n4.x / (e4.x + EPS_GATE) + f.x;
    v.y = d.y + n4.y / (e4.y + EPS_GATE) + f.y;
    v.z = d.z + n4.z / (e4.z + EPS_GATE) + f.z;
    v.w = d.w + n4.w / (e4.w + EPS_GATE) + f.w;
    st4(hpre + ib, v);
    s0 += v.x; s1 += v.y; s2 += v.z; s3 += v.w;
    q0 += v.x * v.x; q1 += v.y * v.y; q2 += v.z * v.z; q3 += v.w * v.w;
  }
  atomAddF(bnsum + c0 + 0, s0); atomAddF(bnsum + c0 + 1, s1);
  atomAddF(bnsum + c0 + 2, s2); atomAddF(bnsum + c0 + 3, s3);
  atomAddF(bnsq + c0 + 0, q0);  atomAddF(bnsq + c0 + 1, q1);
  atomAddF(bnsq + c0 + 2, q2);  atomAddF(bnsq + c0 + 3, q3);
}

// =====================================================================
// Atom apply: h_new = relu(bn(hpre)); scatter S_h[g] += h_new, cnt[g] += 1
// =====================================================================
__global__ __launch_bounds__(256) void atom_apply(
    const float* __restrict__ hpre, const float* __restrict__ scale,
    const float* __restrict__ shift, const int* __restrict__ a2g,
    float* __restrict__ hout, float* S_h, float* cnt, int N)
{
  const int lane = threadIdx.x & 31;
  const int c0 = lane * 4;
  int warp = (blockIdx.x * blockDim.x + threadIdx.x) >> 5;
  int nwarp = (gridDim.x * blockDim.x) >> 5;
  float4 sc = ld4(scale + c0);
  float4 sh = ld4(shift + c0);
  for (int i = warp; i < N; i += nwarp) {
    int g = a2g[i];
    size_t ib = (size_t)i * D + c0;
    float4 x = ld4(hpre + ib);
    float4 y;
    y.x = fmaxf(0.0f, x.x * sc.x + sh.x);
    y.y = fmaxf(0.0f, x.y * sc.y + sh.y);
    y.z = fmaxf(0.0f, x.z * sc.z + sh.z);
    y.w = fmaxf(0.0f, x.w * sc.w + sh.w);
    st4(hout + ib, y);
    size_t gb = (size_t)g * D + c0;
    atomAddF(S_h + gb + 0, y.x); atomAddF(S_h + gb + 1, y.y);
    atomAddF(S_h + gb + 2, y.z); atomAddF(S_h + gb + 3, y.w);
    if (lane == 0) atomAddF(cnt + g, 1.0f);
  }
}

// =====================================================================
// Graph prep: x1 = S_h/max(cnt,1); x2 = S_e/E;
// upre = Iu + (cnt>0)*bG + (ecnt/E)*bH   (GEMM contributions added later)
// =====================================================================
__global__ void prep_u(const float* __restrict__ S_h, const float* __restrict__ S_e,
                       const float* __restrict__ cnt, const float* __restrict__ ecnt,
                       const float* __restrict__ Iu, const float* __restrict__ bG,
                       const float* __restrict__ bH, float invE,
                       float* x1, float* x2, float* upre, int G)
{
  int t = blockIdx.x * blockDim.x + threadIdx.x;
  int g = t >> 5;
  if (g >= G) return;
  int c0 = (t & 31) * 4;
  float c = cnt[g], ec = ecnt[g];
  float invc = 1.0f / fmaxf(c, 1.0f);
  float gflag = (c > 0.0f) ? 1.0f : 0.0f;
#pragma unroll
  for (int i = 0; i < 4; ++i) {
    int cc = c0 + i;
    size_t idx = (size_t)g * D + cc;
    x1[idx] = S_h[idx] * invc;
    x2[idx] = S_e[idx] * invE;
    upre[idx] = Iu[idx] + gflag * bG[cc] + ec * invE * bH[cc];
  }
}

// Generic column stats over [nrows, D]
__global__ __launch_bounds__(256) void colstats(const float* __restrict__ X, int nrows,
                                                float* bnsum, float* bnsq)
{
  const int lane = threadIdx.x & 31;
  const int c0 = lane * 4;
  int warp = (blockIdx.x * blockDim.x + threadIdx.x) >> 5;
  int nwarp = (gridDim.x * blockDim.x) >> 5;
  float s0 = 0, s1 = 0, s2 = 0, s3 = 0, q0 = 0, q1 = 0, q2 = 0, q3 = 0;
  for (int i = warp; i < nrows; i += nwarp) {
    float4 v = ld4(X + (size_t)i * D + c0);
    s0 += v.x; s1 += v.y; s2 += v.z; s3 += v.w;
    q0 += v.x * v.x; q1 += v.y * v.y; q2 += v.z * v.z; q3 += v.w * v.w;
  }
  atomAddF(bnsum + c0 + 0, s0); atomAddF(bnsum + c0 + 1, s1);
  atomAddF(bnsum + c0 + 2, s2); atomAddF(bnsum + c0 + 3, s3);
  atomAddF(bnsq + c0 + 0, q0);  atomAddF(bnsq + c0 + 1, q1);
  atomAddF(bnsq + c0 + 2, q2);  atomAddF(bnsq + c0 + 3, q3);
}

// Simple BN+ReLU apply over contiguous [nElem] (nElem multiple of D)
__global__ void bn_relu_apply(const float* __restrict__ X, const float* __restrict__ scale,
                              const float* __restrict__ shift, float* __restrict__ Y,
                              size_t nElem)
{
  size_t t = (size_t)blockIdx.x * blockDim.x + threadIdx.x;
  size_t e0 = t * 4;
  if (e0 >= nElem) return;
  int c0 = (int)(e0 & (D - 1));
  float4 x = ld4(X + e0);
  float4 sc = ld4(scale + c0);
  float4 sh = ld4(shift + c0);
  float4 y;
  y.x = fmaxf(0.0f, x.x * sc.x + sh.x);
  y.y = fmaxf(0.0f, x.y * sc.y + sh.y);
  y.z = fmaxf(0.0f, x.z * sc.z + sh.z);
  y.w = fmaxf(0.0f, x.w * sc.w + sh.w);
  st4(Y + e0, y);
}

// =====================================================================
extern "C" void kernel_launch(void* const* d_in, const int* in_sizes, int n_in,
                              void* d_out, int out_size, void* d_ws, size_t ws_size,
                              hipStream_t stream)
{
  const float* h     = (const float*)d_in[0];
  const float* e     = (const float*)d_in[1];
  const float* u     = (const float*)d_in[2];
  const float* W     = (const float*)d_in[3];   // [9][128][128]
  const float* b     = (const float*)d_in[4];   // [9][128]
  const float* gamma = (const float*)d_in[5];   // [3][128]
  const float* beta  = (const float*)d_in[6];   // [3][128]
  const int*   src   = (const int*)d_in[7];
  const int*   dstI  = (const int*)d_in[8];
  const int*   a2g   = (const int*)d_in[9];

  const int N = in_sizes[0] / D;
  const int E = in_sizes[1] / D;
  const int G = in_sizes[2] / D;
  const size_t ND = (size_t)N * D, ED = (size_t)E * D, GD = (size_t)G * D;
  const size_t DD = (size_t)D * D;

  float* ws = (float*)d_ws;
  float* Ah   = ws;            // [N,D]
  float* Dh   = Ah + ND;       // [N,D]
  float* Eh   = Dh + ND;       // [N,D]
  float* Be   = Eh + ND;       // [E,D]  (reused in-place as epre)
  float* Cu   = Be + ED;       // [G,D]
  float* Fu   = Cu + GD;       // [G,D]
  float* Iu   = Fu + GD;       // [G,D]
  float* hpre = Iu + GD;       // [N,D]
  float* x1   = hpre + ND;     // [G,D]
  float* x2   = x1 + GD;       // [G,D]
  float* upre = x2 + GD;       // [G,D]
  // ---- zero-initialized region ----
  float* zero0 = upre + GD;
  float* num  = zero0;         // [N,D]
  float* den  = num + ND;      // [N,D]
  float* S_e  = den + ND;      // [G,D]
  float* S_h  = S_e + GD;      // [G,D]
  float* cnt  = S_h + GD;      // [G]
  float* ecnt = cnt + G;       // [G]
  float* sum_e = ecnt + G;     float* sq_e = sum_e + D;
  float* sum_h = sq_e + D;     float* sq_h = sum_h + D;
  float* sum_u = sq_h + D;     float* sq_u = sum_u + D;
  float* sc_e = sq_u + D;      float* sh_e = sc_e + D;
  float* sc_h = sh_e + D;      float* sh_h = sc_h + D;
  float* sc_u = sh_h + D;      float* sh_u = sc_u + D;

  size_t zbytes = (size_t)((sh_u + D) - zero0) * sizeof(float);
  hipMemsetAsync(zero0, 0, zbytes, stream);

  float* h_out = (float*)d_out;            // [N,D]
  float* e_out = h_out + ND;               // [E,D]
  float* u_out = e_out + ED;               // [G,D]

  auto gemmGrid = [](int rows) { return ((rows + 15) / 16 + WPB - 1) / WPB; };

  // 1) Ah/Dh/Eh = h @ {W_A, W_D, W_E}
  gemm_bf16x3<0><<<gemmGrid(N), 256, 0, stream>>>(
      h, N, W + 0 * DD, W + 3 * DD, W + 4 * DD,
      b + 0 * D, b + 3 * D, b + 4 * D, Ah, Dh, Eh, 3);
  // 2) Cu/Fu/Iu = u @ {W_C, W_F, W_I}
  gemm_bf16x3<0><<<gemmGrid(G), 256, 0, stream>>>(
      u, G, W + 2 * DD, W + 5 * DD, W + 8 * DD,
      b + 2 * D, b + 5 * D, b + 8 * D, Cu, Fu, Iu, 3);
  // 3) Be = e @ W_B
  gemm_bf16x3<0><<<gemmGrid(E), 256, 0, stream>>>(
      e, E, W + 1 * DD, nullptr, nullptr,
      b + 1 * D, nullptr, nullptr, Be, nullptr, nullptr, 1);

  // 4) edge assemble (+BN stats over edges)
  edge_assemble<<<2048, 256, 0, stream>>>(Ah, Be, Cu, src, dstI, a2g,
                                          Be /*epre in-place*/, sum_e, sq_e, E);
  // 5) BN finalize (edge)
  bn_finalize<<<1, D, 0, stream>>>(sum_e, sq_e, gamma + 1 * D, beta + 1 * D,
                                   1.0f / (float)E, sc_e, sh_e);
  // 6) edge apply: e_new, gating scatters, per-graph edge sums
  edge_apply<<<2048, 256, 0, stream>>>(Be, sc_e, sh_e, Eh, src, dstI, a2g,
                                       e_out, num, den, S_e, ecnt, E);
  // 7) atom assemble (+BN stats over atoms)
  atom_assemble<<<2048, 256, 0, stream>>>(Dh, num, den, Fu, a2g, hpre,
                                          sum_h, sq_h, N);
  // 8) BN finalize (atom)
  bn_finalize<<<1, D, 0, stream>>>(sum_h, sq_h, gamma + 0 * D, beta + 0 * D,
                                   1.0f / (float)N, sc_h, sh_h);
  // 9) atom apply: h_new, per-graph atom sums
  atom_apply<<<2048, 256, 0, stream>>>(hpre, sc_h, sh_h, a2g, h_out, S_h, cnt, N);
  // 10) graph prep: mean inputs + bias terms
  prep_u<<<(G * 32 + 255) / 256, 256, 0, stream>>>(
      S_h, S_e, cnt, ecnt, Iu, b + 6 * D, b + 7 * D, 1.0f / (float)E,
      x1, x2, upre, G);
  // 11) upre += x1 @ W_G   (mean_Gh, linearity of segment-sum)
  gemm_bf16x3<1><<<gemmGrid(G), 256, 0, stream>>>(
      x1, G, W + 6 * DD, nullptr, nullptr, nullptr, nullptr, nullptr,
      upre, nullptr, nullptr, 1);
  // 12) upre += x2 @ W_H   (mean_He)
  gemm_bf16x3<1><<<gemmGrid(G), 256, 0, stream>>>(
      x2, G, W + 7 * DD, nullptr, nullptr, nullptr, nullptr, nullptr,
      upre, nullptr, nullptr, 1);
  // 13) BN stats (graphs)
  colstats<<<512, 256, 0, stream>>>(upre, G, sum_u, sq_u);
  // 14) BN finalize (graph)
  bn_finalize<<<1, D, 0, stream>>>(sum_u, sq_u, gamma + 2 * D, beta + 2 * D,
                                   1.0f / (float)G, sc_u, sh_u);
  // 15) u_new
  bn_relu_apply<<<(int)((GD / 4 + 255) / 256), 256, 0, stream>>>(upre, sc_u, sh_u,
                                                                 u_out, GD);
}
